// Wave_57389353009773
// MI455X (gfx1250) — matrix operand order repaired and checked by reading, not compiled, not run on hardware
//
#include <hip/hip_runtime.h>
#include <hip/hip_bf16.h>

// ---------------------------------------------------------------------------
// Types / constants
// ---------------------------------------------------------------------------
typedef __bf16 bf16_t;
typedef __attribute__((ext_vector_type(16))) __bf16 v16bf;
typedef __attribute__((ext_vector_type(8)))  float  v8f;
typedef __attribute__((ext_vector_type(4)))  int    v4i;

constexpr int kD  = 1024;   // embed dim
constexpr int kH  = 16;     // heads
constexpr int kDh = 64;     // head dim

union PK4 { uint2 u; bf16_t h[4]; };
union PK8 { uint4 u; bf16_t h[8]; };

// Optional gfx1250 async-to-LDS staging (guarded; falls back to VGPR path)
#if __has_builtin(__builtin_amdgcn_global_load_async_to_lds_b128) && \
    __has_builtin(__builtin_amdgcn_s_wait_asynccnt)
#define GEMM_ASYNC_LDS 1
// builtin signature (from hipcc diagnostic): (v4i __device__*, v4i __shared__*, imm, imm)
typedef __attribute__((address_space(1))) v4i* gbl_v4i_p;
typedef __attribute__((address_space(3))) v4i* lds_v4i_p;
#else
#define GEMM_ASYNC_LDS 0
#endif

// ---------------------------------------------------------------------------
// WMMA helpers (CDNA5 layouts from CDNA5 ISA)
// A (16x32 bf16): lane m=lane&15, kb=(lane>>4)*8; elems 0..7 -> K=kb..kb+7,
//                 elems 8..15 -> K=16+kb..16+kb+7.   (row-major contiguous)
// B (32x16 bf16): lane n=lane&15, kb=(lane>>4)*8; same K order as A.
//   -> stored K-contiguous per n-row ("B^T row-major") so loads vectorize.
// C/D (16x16 f32): lane n=lane&15, mb=(lane>>4)*8; vgpr v -> row mb+v.
// ---------------------------------------------------------------------------
__device__ inline v8f wmma_bf16(v16bf a, v16bf b, v8f c) {
  return __builtin_amdgcn_wmma_f32_16x16x32_bf16(false, a, false, b, (short)0,
                                                 c, false, false);
}

// A fragment from row-major [16 x stride] tile (K=kb..kb+7, 16+kb..16+kb+7 per lane)
__device__ inline v16bf load_a_frag(const bf16_t* sm, int stride, int lane) {
  int m  = lane & 15;
  int kb = (lane >> 4) << 3;
  v16bf a;
#pragma unroll
  for (int i = 0; i < 8; ++i) a[i]     = sm[m * stride + kb + i];
#pragma unroll
  for (int i = 0; i < 8; ++i) a[8 + i] = sm[m * stride + 16 + kb + i];
  return a;
}

// B fragment from K-contiguous layout rows[n][k]: K=kb..kb+7, 16+kb..16+kb+7 per lane
__device__ inline v16bf load_row_frag(const bf16_t* sm, int stride, int kofs,
                                      int lane) {
  int n  = lane & 15;
  int kb = (lane >> 4) << 3;
  v16bf b;
#pragma unroll
  for (int i = 0; i < 8; ++i) b[i]     = sm[n * stride + kofs + kb + i];
#pragma unroll
  for (int i = 0; i < 8; ++i) b[8 + i] = sm[n * stride + kofs + 16 + kb + i];
  return b;
}

// A fragment (K = 16 real tokens, upper 16 zero) from Kt[row][token] (stride 16)
__device__ inline v16bf load_kt_frag(const bf16_t* sm, int lane) {
  int m  = lane & 15;
  int kb = (lane >> 4) << 3;
  v16bf a;
#pragma unroll
  for (int i = 0; i < 8; ++i) a[i]     = sm[m * 16 + kb + i];
#pragma unroll
  for (int i = 0; i < 8; ++i) a[8 + i] = (bf16_t)0.0f;
  return a;
}

// ---------------------------------------------------------------------------
// fp32 -> bf16 conversion (plain and transposing variants)
// ---------------------------------------------------------------------------
__global__ void cvt_bf16_kernel(const float* __restrict__ in,
                                bf16_t* __restrict__ out, int n) {
  int i = blockIdx.x * blockDim.x + threadIdx.x;
  if (i < n) out[i] = (bf16_t)in[i];
}

// out[c*R + r] = in[r*C + c]   (weight K x N -> N x K bf16)
__global__ void cvt_bf16_t_kernel(const float* __restrict__ in,
                                  bf16_t* __restrict__ out, int R, int C) {
  int i = blockIdx.x * blockDim.x + threadIdx.x;
  if (i >= R * C) return;
  int c = i % C, r = i / C;
  out[(size_t)c * R + r] = (bf16_t)in[(size_t)r * C + c];
}

// ---------------------------------------------------------------------------
// GEMM: C[M,N] = A[M,K](bf16) @ B[K,N] + bias, fp32 out. B passed as B^T (NxK).
// 256 threads = 8 waves; tile 128x64, BK=32; wave owns a 16x64 strip.
// Tile staging uses gfx1250 async-to-LDS DMA when available.
// ---------------------------------------------------------------------------
#define GEMM_BM 128
#define GEMM_BN 64
#define GEMM_BK 32

__global__ __launch_bounds__(256)
void gemm_bias_kernel(const bf16_t* __restrict__ A,   // M x K row-major
                      const bf16_t* __restrict__ BT,  // N x K row-major
                      const float* __restrict__ bias, float* __restrict__ C,
                      int M, int N, int K) {
  __shared__ __align__(16) bf16_t As[GEMM_BM][GEMM_BK];
  __shared__ __align__(16) bf16_t BsT[GEMM_BN][GEMM_BK];
  const int tid  = threadIdx.x;
  const int lane = tid & 31;
  const int wave = tid >> 5;
  const int bm = blockIdx.y * GEMM_BM;
  const int bn = blockIdx.x * GEMM_BN;

  v8f acc[4] = {};

  // per-thread staging coordinates
  const int arow = tid >> 1;            // 0..127
  const int akc  = (tid & 1) * 16;      // 0 / 16
  const int brow = tid >> 2;            // 0..63
  const int bkc  = (tid & 3) * 8;       // 0,8,16,24

  for (int kt = 0; kt < K; kt += GEMM_BK) {
#if GEMM_ASYNC_LDS
    {   // A tile 128x32: 2x async b128 per thread; B^T tile 64x32: 1x b128
      const bf16_t* asrc = A + (size_t)(bm + arow) * K + kt + akc;
      __builtin_amdgcn_global_load_async_to_lds_b128(
          (gbl_v4i_p)asrc, (lds_v4i_p)&As[arow][akc], 0, 0);
      __builtin_amdgcn_global_load_async_to_lds_b128(
          (gbl_v4i_p)(asrc + 8), (lds_v4i_p)&As[arow][akc + 8], 0, 0);
      const bf16_t* bsrc = BT + (size_t)(bn + brow) * K + kt + bkc;
      __builtin_amdgcn_global_load_async_to_lds_b128(
          (gbl_v4i_p)bsrc, (lds_v4i_p)&BsT[brow][bkc], 0, 0);
      __builtin_amdgcn_s_wait_asynccnt(0);
    }
#else
    {   // A tile 128x32: 2x uint4 per thread (through VGPRs)
      const uint4* src = (const uint4*)(A + (size_t)(bm + arow) * K + kt + akc);
      uint4* dst = (uint4*)(&As[arow][akc]);
      dst[0] = src[0];
      dst[1] = src[1];
      if (kt + GEMM_BK < K)
        __builtin_prefetch(A + (size_t)(bm + arow) * K + kt + GEMM_BK + akc,
                           0, 3);
    }
    {   // B^T tile 64x32: 1x uint4 per thread
      *(uint4*)(&BsT[brow][bkc]) =
          *(const uint4*)(BT + (size_t)(bn + brow) * K + kt + bkc);
      if (kt + GEMM_BK < K)
        __builtin_prefetch(BT + (size_t)(bn + brow) * K + kt + GEMM_BK + bkc,
                           0, 3);
    }
#endif
    __syncthreads();
    v16bf af = load_a_frag(&As[wave * 16][0], GEMM_BK, lane);
#pragma unroll
    for (int nt = 0; nt < 4; ++nt) {
      v16bf bfr = load_row_frag(&BsT[nt * 16][0], GEMM_BK, 0, lane);
      acc[nt] = wmma_bf16(af, bfr, acc[nt]);
    }
    __syncthreads();
  }

  int n0 = lane & 15;
  int mb = (lane >> 4) * 8;
#pragma unroll
  for (int nt = 0; nt < 4; ++nt) {
    int col = bn + nt * 16 + n0;
    float bv = bias ? bias[col] : 0.0f;
#pragma unroll
    for (int v = 0; v < 8; ++v) {
      int row = bm + wave * 16 + mb + v;
      C[(size_t)row * N + col] = acc[nt][v] + bv;
    }
  }
}

// ---------------------------------------------------------------------------
// Gate scalars: g = sigmoid(x_head . Wg + bg)  -> log(clip(g,1e-6))
//               R = silu(K_head . Ws + bs)
// Layout of Glog/Rbuf/Gc: [(b*H+h)*L + l]
// ---------------------------------------------------------------------------
__global__ void gates1_kernel(const float* __restrict__ x,
                              const float* __restrict__ K1,
                              const float* __restrict__ Wg, const float* bg,
                              const float* __restrict__ Wsv, const float* bs,
                              float* __restrict__ Glog, float* __restrict__ Rbuf,
                              int Bn, int L) {
  int idx = blockIdx.x * blockDim.x + threadIdx.x;
  int total = Bn * L * kH;
  if (idx >= total) return;
  int h = idx % kH;
  int l = (idx / kH) % L;
  int b = idx / (kH * L);
  size_t base = ((size_t)(b * L + l) * kD) + (size_t)h * kDh;
  float ag = bg[0], as = bs[0];
  for (int k = 0; k < kDh; ++k) {
    ag += x[base + k]  * Wg[k];
    as += K1[base + k] * Wsv[k];
  }
  float g = 1.0f / (1.0f + expf(-ag));
  size_t o = (size_t)(b * kH + h) * L + l;
  Glog[o] = logf(fmaxf(g, 1e-6f));
  Rbuf[o] = as / (1.0f + expf(-as));   // silu
}

__global__ void gates2_kernel(const float* __restrict__ Glog,
                              float* __restrict__ Gc, int nStream, int L) {
  int s = blockIdx.x * blockDim.x + threadIdx.x;
  if (s >= nStream) return;
  float c = 0.0f;
  for (int l = 0; l < L; ++l) {
    c += Glog[(size_t)s * L + l];
    float cl = fminf(fmaxf(c, -30.0f), 30.0f);
    Gc[(size_t)s * L + l] = expf(cl) + 1e-6f;
  }
}

// ---------------------------------------------------------------------------
// GLA chunked scan. One WG per (b,h), chunk = 16 tokens, 8 waves.
// fp32 state lives in registers: wave w owns D-tiles (w*2, w*2+1) of the
// 64x64 state; a bf16 transposed mirror Stt[n][kf] serves as WMMA B operand.
//   O1_l = (Gc_l/Gc_p) Q_l @ S + sum_{j<=l} (Gc_l/Gc_j) R_j (Q_l.K_j) V_j
//   S   <- (Gc_e/Gc_p) S + sum_j (Gc_e/Gc_j) R_j K_j^T V_j
// ---------------------------------------------------------------------------
__global__ __launch_bounds__(256)
void gla_scan_kernel(const float* __restrict__ Qg, const float* __restrict__ Kg,
                     const float* __restrict__ Xg, const float* __restrict__ Gc,
                     const float* __restrict__ Rg, float* __restrict__ O1,
                     int Bn, int L) {
  const int bh = blockIdx.x;
  const int b  = bh / kH;
  const int h  = bh % kH;
  const int tid  = threadIdx.x;
  const int lane = tid & 31;
  const int wave = tid >> 5;

  __shared__ __align__(16) bf16_t Qs[16][64];   // decay-scaled Q (A-op)
  __shared__ __align__(16) bf16_t Qr[16][64];   // raw Q (A-op)
  __shared__ __align__(16) bf16_t Ks[16][64];   // raw K (B-op for P: [tok][kf])
  __shared__ __align__(16) bf16_t Stt[64][64];  // S^T mirror: [n][kf]
  __shared__ __align__(16) bf16_t Vt[64][32];   // V^T: [feat][tok], tok 16..31 = 0
  __shared__ __align__(16) bf16_t Kdt[64][16];  // (R*decay*K)^T: [feat][tok]
  __shared__ __align__(16) bf16_t Pb[16][32];   // P row-major, cols 16..31 = 0
  __shared__ float gcs[17];                     // [0]=Gc_prev, [1+j]=chunk Gc
  __shared__ float Rs[16];

  for (int i = tid; i < 64 * 64; i += 256) (&Stt[0][0])[i] = (bf16_t)0.0f;
  for (int i = tid; i < 64 * 16; i += 256) Vt[i >> 4][16 + (i & 15)] = (bf16_t)0.0f;
  for (int i = tid; i < 16 * 16; i += 256) Pb[i >> 4][16 + (i & 15)] = (bf16_t)0.0f;
  __syncthreads();

  v8f sreg[2] = {};                 // fp32 state tiles owned by this wave

  const int j  = tid >> 4;          // token in chunk
  const int c0 = (tid & 15) * 4;    // feature group

  for (int t0 = 0; t0 < L; t0 += 16) {
    // ---- phase 1: load chunk, stage Qr/Ks row-major and V transposed ----
    float qv[4], kv[4];
    {
      size_t base = ((size_t)(b * L + t0 + j) * kD) + (size_t)h * kDh + c0;
      PK4 pq, pk;
#pragma unroll
      for (int u = 0; u < 4; ++u) {
        qv[u] = Qg[base + u];
        kv[u] = Kg[base + u];
        float vv = Xg[base + u];
        pq.h[u] = (bf16_t)qv[u];
        pk.h[u] = (bf16_t)kv[u];
        Vt[c0 + u][j] = (bf16_t)vv;
      }
      *(uint2*)&Qr[j][c0] = pq.u;
      *(uint2*)&Ks[j][c0] = pk.u;
      if (tid < 16) {
        gcs[1 + tid] = Gc[(size_t)bh * L + t0 + tid];
        Rs[tid]      = Rg[(size_t)bh * L + t0 + tid];
      }
      if (tid == 16) gcs[0] = (t0 == 0) ? 1.0f : Gc[(size_t)bh * L + t0 - 1];
    }
    __syncthreads();
    // ---- phase 1b: scaled Q (row-major) and decayed K (transposed) ----
    {
      float qscale = gcs[1 + j] / gcs[0];
      float kscale = Rs[j] * (gcs[16] / gcs[1 + j]);
      PK4 ps;
#pragma unroll
      for (int u = 0; u < 4; ++u) {
        ps.h[u] = (bf16_t)(qv[u] * qscale);
        Kdt[c0 + u][j] = (bf16_t)(kv[u] * kscale);
      }
      *(uint2*)&Qs[j][c0] = ps.u;
    }
    __syncthreads();

    // ---- phase 2: waves 0-3 -> O_state = Qs @ S ; wave 4 -> P ----
    v8f oacc = {};
    if (wave < 4) {
      int nt = wave;
#pragma unroll
      for (int ks = 0; ks < 2; ++ks) {
        v16bf af  = load_a_frag(&Qs[0][ks * 32], 64, lane);
        v16bf bfr = load_row_frag(&Stt[nt * 16][0], 64, ks * 32, lane);
        oacc = wmma_bf16(af, bfr, oacc);
      }
    } else if (wave == 4) {
      v8f pacc = {};
#pragma unroll
      for (int ks = 0; ks < 2; ++ks) {
        v16bf af  = load_a_frag(&Qr[0][ks * 32], 64, lane);
        v16bf bfr = load_row_frag(&Ks[0][0], 64, ks * 32, lane);  // K^T
        pacc = wmma_bf16(af, bfr, pacc);
      }
      int n  = lane & 15;             // column = source token
      int mb = (lane >> 4) * 8;
#pragma unroll
      for (int v = 0; v < 8; ++v) {
        int row = mb + v;             // dest token
        float pv = (n <= row)
                       ? pacc[v] * Rs[n] * (gcs[1 + row] / gcs[1 + n])
                       : 0.0f;
        Pb[row][n] = (bf16_t)pv;
      }
    }
    __syncthreads();

    // ---- phase 3: O_attn + store; all waves update their state tiles ----
    if (wave < 4) {
      int nt = wave;
      v16bf af  = load_a_frag(&Pb[0][0], 32, lane);
      v16bf bfr = load_row_frag(&Vt[nt * 16][0], 32, 0, lane);
      oacc = wmma_bf16(af, bfr, oacc);
      int n  = lane & 15;
      int mb = (lane >> 4) * 8;
#pragma unroll
      for (int v = 0; v < 8; ++v) {
        int row = mb + v;
        O1[((size_t)(b * L + t0 + row) * kD) + (size_t)h * kDh + nt * 16 + n] =
            oacc[v];
      }
    }
    {
      float tot = gcs[16] / gcs[0];
#pragma unroll
      for (int tt = 0; tt < 2; ++tt) {
        int tile = wave * 2 + tt;
        int mt = tile >> 2, nt = tile & 3;
        v16bf af  = load_kt_frag(&Kdt[mt * 16][0], lane);   // K^T (decayed)
        v16bf bfr = load_row_frag(&Vt[nt * 16][0], 32, 0, lane);
        v8f cc;
#pragma unroll
        for (int v = 0; v < 8; ++v) cc[v] = tot * sreg[tt][v];
        sreg[tt] = wmma_bf16(af, bfr, cc);
      }
    }
    __syncthreads();   // everyone done reading old Stt (phase 2)

    // ---- phase 4: refresh bf16 transposed mirror (one b128 per tile) ----
#pragma unroll
    for (int tt = 0; tt < 2; ++tt) {
      int tile = wave * 2 + tt;
      int mt = tile >> 2, nt = tile & 3;
      int n  = lane & 15;
      int mb = (lane >> 4) * 8;
      PK8 pk;
#pragma unroll
      for (int v = 0; v < 8; ++v) pk.h[v] = (bf16_t)sreg[tt][v];
      *(uint4*)&Stt[nt * 16 + n][mt * 16 + mb] = pk.u;
    }
    __syncthreads();
  }
}

// ---------------------------------------------------------------------------
// MA chunked scan (decay = 1): S += k2^T E; O2_l = q2_l @ S (inclusive).
// q2 = -leaky_relu(-Q/8, 0.02); k2 = sigmoid(K2pre * 0.02/32); E = V[l+1]-O1[l]
// Writes Osum[l+1] = O1[l+1] + O2_l ; Osum[0] = O1[0].
// ---------------------------------------------------------------------------
__global__ __launch_bounds__(256)
void ma_scan_kernel(const float* __restrict__ Qg, const float* __restrict__ K2g,
                    const float* __restrict__ Xg, const float* __restrict__ O1,
                    float* __restrict__ Osum, int Bn, int L) {
  const int bh = blockIdx.x;
  const int b  = bh / kH;
  const int h  = bh % kH;
  const int tid  = threadIdx.x;
  const int lane = tid & 31;
  const int wave = tid >> 5;

  __shared__ __align__(16) bf16_t Q2s[16][64];  // q2 (A-op)
  __shared__ __align__(16) bf16_t K2s[16][64];  // k2 (B-op for P: [tok][kf])
  __shared__ __align__(16) bf16_t Stt[64][64];  // S^T mirror
  __shared__ __align__(16) bf16_t Et[64][32];   // E^T: [feat][tok], pad = 0
  __shared__ __align__(16) bf16_t K2t[64][16];  // k2^T: [feat][tok]
  __shared__ __align__(16) bf16_t Pb[16][32];   // cols 16..31 = 0

  for (int i = tid; i < 64 * 64; i += 256) (&Stt[0][0])[i] = (bf16_t)0.0f;
  for (int i = tid; i < 64 * 16; i += 256) Et[i >> 4][16 + (i & 15)] = (bf16_t)0.0f;
  for (int i = tid; i < 16 * 16; i += 256) Pb[i >> 4][16 + (i & 15)] = (bf16_t)0.0f;
  __syncthreads();

  v8f sreg[2] = {};

  const int j  = tid >> 4;
  const int c0 = (tid & 15) * 4;
  const int Lm = L - 1;   // MA token count

  for (int t0 = 0; t0 < Lm; t0 += 16) {
    const int valid = (Lm - t0 < 16) ? (Lm - t0) : 16;
    {
      size_t base = ((size_t)(b * L + t0 + j) * kD) + (size_t)h * kDh + c0;
      PK4 pq, pk;
#pragma unroll
      for (int u = 0; u < 4; ++u) {
        float q2v = 0.0f, k2v = 0.0f, ev = 0.0f;
        if (j < valid) {
          float q = Qg[base + u];
          q2v = (q >= 0.0f) ? 0.0025f * q : 0.125f * q;   // 0.02/8 : 1/8
          float kp = K2g[base + u];
          k2v = 1.0f / (1.0f + expf(-kp * 0.000625f));     // 0.02/32
          ev  = Xg[base + kD + u] - O1[base + u];          // V[l+1]-O1[l]
        }
        pq.h[u] = (bf16_t)q2v;
        pk.h[u] = (bf16_t)k2v;
        Et[c0 + u][j]  = (bf16_t)ev;
        K2t[c0 + u][j] = pk.h[u];
        if (t0 == 0 && j == 0) {  // Osum[0] = O1[0]
          size_t b0 = ((size_t)(b * L) * kD) + (size_t)h * kDh + c0;
          Osum[b0 + u] = O1[b0 + u];
        }
      }
      *(uint2*)&Q2s[j][c0] = pq.u;
      *(uint2*)&K2s[j][c0] = pk.u;
    }
    __syncthreads();

    v8f oacc = {};
    if (wave < 4) {            // O_state = q2 @ S
      int nt = wave;
#pragma unroll
      for (int ks = 0; ks < 2; ++ks) {
        v16bf af  = load_a_frag(&Q2s[0][ks * 32], 64, lane);
        v16bf bfr = load_row_frag(&Stt[nt * 16][0], 64, ks * 32, lane);
        oacc = wmma_bf16(af, bfr, oacc);
      }
    } else if (wave == 4) {    // P = q2 @ k2^T, causal mask
      v8f pacc = {};
#pragma unroll
      for (int ks = 0; ks < 2; ++ks) {
        v16bf af  = load_a_frag(&Q2s[0][ks * 32], 64, lane);
        v16bf bfr = load_row_frag(&K2s[0][0], 64, ks * 32, lane);
        pacc = wmma_bf16(af, bfr, pacc);
      }
      int n  = lane & 15;
      int mb = (lane >> 4) * 8;
#pragma unroll
      for (int v = 0; v < 8; ++v) {
        int row = mb + v;
        Pb[row][n] = (bf16_t)((n <= row) ? pacc[v] : 0.0f);
      }
    }
    __syncthreads();

    if (wave < 4) {            // O2 = O_state + P @ E ; store shifted
      int nt = wave;
      v16bf af  = load_a_frag(&Pb[0][0], 32, lane);
      v16bf bfr = load_row_frag(&Et[nt * 16][0], 32, 0, lane);
      oacc = wmma_bf16(af, bfr, oacc);
      int n  = lane & 15;
      int mb = (lane >> 4) * 8;
#pragma unroll
      for (int v = 0; v < 8; ++v) {
        int row = mb + v;
        if (row < valid) {
          size_t o = ((size_t)(b * L + t0 + row + 1) * kD) +
                     (size_t)h * kDh + nt * 16 + n;
          Osum[o] = O1[o] + oacc[v];
        }
      }
    }
#pragma unroll
    for (int tt = 0; tt < 2; ++tt) {   // S += k2^T @ E
      int tile = wave * 2 + tt;
      int mt = tile >> 2, nt = tile & 3;
      v16bf af  = load_kt_frag(&K2t[mt * 16][0], lane);
      v16bf bfr = load_row_frag(&Et[nt * 16][0], 32, 0, lane);
      sreg[tt] = wmma_bf16(af, bfr, sreg[tt]);
    }
    __syncthreads();

#pragma unroll
    for (int tt = 0; tt < 2; ++tt) {   // refresh mirror
      int tile = wave * 2 + tt;
      int mt = tile >> 2, nt = tile & 3;
      int n  = lane & 15;
      int mb = (lane >> 4) * 8;
      PK8 pk;
#pragma unroll
      for (int v = 0; v < 8; ++v) pk.h[v] = (bf16_t)sreg[tt][v];
      *(uint4*)&Stt[nt * 16 + n][mt * 16 + mb] = pk.u;
    }
    __syncthreads();
  }
}

// ---------------------------------------------------------------------------
// Host launcher
// ---------------------------------------------------------------------------
extern "C" void kernel_launch(void* const* d_in, const int* in_sizes, int n_in,
                              void* d_out, int out_size, void* d_ws, size_t ws_size,
                              hipStream_t stream) {
  const float* x   = (const float*)d_in[0];
  const float* Wq  = (const float*)d_in[1];
  const float* bq  = (const float*)d_in[2];
  const float* Wk1 = (const float*)d_in[3];
  const float* bk1 = (const float*)d_in[4];
  const float* Wk2 = (const float*)d_in[5];
  const float* bk2 = (const float*)d_in[6];
  const float* Wg  = (const float*)d_in[7];
  const float* bg  = (const float*)d_in[8];
  const float* Wsv = (const float*)d_in[9];
  const float* bs  = (const float*)d_in[10];
  const float* Wp  = (const float*)d_in[11];
  const float* bp  = (const float*)d_in[12];
  float* out = (float*)d_out;

  const int D  = kD;
  const int L  = 1024;
  const int BL = in_sizes[0] / D;     // B*L = 2048
  const int Bn = BL / L;              // 2
  const int M  = BL;

  char* ws = (char*)d_ws;
  size_t off = 0;
  auto alloc = [&](size_t bytes) {
    void* p = ws + off;
    off = (off + bytes + 255) & ~(size_t)255;
    return p;
  };
  bf16_t* xb    = (bf16_t*)alloc((size_t)M * D * sizeof(bf16_t));
  bf16_t* WqbT  = (bf16_t*)alloc((size_t)D * D * sizeof(bf16_t));
  bf16_t* Wk1bT = (bf16_t*)alloc((size_t)D * D * sizeof(bf16_t));
  bf16_t* Wk2bT = (bf16_t*)alloc((size_t)D * D * sizeof(bf16_t));
  bf16_t* WpbT  = (bf16_t*)alloc((size_t)D * D * sizeof(bf16_t));
  float*  Qbuf  = (float*)alloc((size_t)M * D * sizeof(float));
  float*  K1buf = (float*)alloc((size_t)M * D * sizeof(float));
  float*  K2buf = (float*)alloc((size_t)M * D * sizeof(float));
  float*  O1buf = (float*)alloc((size_t)M * D * sizeof(float));
  float*  Osum  = (float*)alloc((size_t)M * D * sizeof(float));
  bf16_t* Osumb = (bf16_t*)alloc((size_t)M * D * sizeof(bf16_t));
  float*  Glog  = (float*)alloc((size_t)Bn * kH * L * sizeof(float));
  float*  Gc    = (float*)alloc((size_t)Bn * kH * L * sizeof(float));
  float*  Rbuf  = (float*)alloc((size_t)Bn * kH * L * sizeof(float));

  // 1) bf16 conversions (weights transposed for vectorized B-operand staging)
  int nX = M * D, nW = D * D;
  cvt_bf16_kernel<<<(nX + 255) / 256, 256, 0, stream>>>(x, xb, nX);
  cvt_bf16_t_kernel<<<(nW + 255) / 256, 256, 0, stream>>>(Wq,  WqbT,  D, D);
  cvt_bf16_t_kernel<<<(nW + 255) / 256, 256, 0, stream>>>(Wk1, Wk1bT, D, D);
  cvt_bf16_t_kernel<<<(nW + 255) / 256, 256, 0, stream>>>(Wk2, Wk2bT, D, D);
  cvt_bf16_t_kernel<<<(nW + 255) / 256, 256, 0, stream>>>(Wp,  WpbT,  D, D);

  // 2) projections
  dim3 gg(D / GEMM_BN, M / GEMM_BM);
  gemm_bias_kernel<<<gg, 256, 0, stream>>>(xb, WqbT,  bq,  Qbuf,  M, D, D);
  gemm_bias_kernel<<<gg, 256, 0, stream>>>(xb, Wk1bT, bk1, K1buf, M, D, D);
  gemm_bias_kernel<<<gg, 256, 0, stream>>>(xb, Wk2bT, bk2, K2buf, M, D, D);

  // 3) gates
  int nG = Bn * L * kH;
  gates1_kernel<<<(nG + 255) / 256, 256, 0, stream>>>(x, K1buf, Wg, bg, Wsv, bs,
                                                      Glog, Rbuf, Bn, L);
  gates2_kernel<<<1, 32, 0, stream>>>(Glog, Gc, Bn * kH, L);

  // 4) GLA scan (WMMA)
  gla_scan_kernel<<<Bn * kH, 256, 0, stream>>>(Qbuf, K1buf, x, Gc, Rbuf,
                                               O1buf, Bn, L);

  // 5) MA scan (WMMA) -> Osum = O1 + shifted O2
  ma_scan_kernel<<<Bn * kH, 256, 0, stream>>>(Qbuf, K2buf, x, O1buf, Osum,
                                              Bn, L);

  // 6) output projection
  cvt_bf16_kernel<<<(nX + 255) / 256, 256, 0, stream>>>(Osum, Osumb, nX);
  gemm_bias_kernel<<<gg, 256, 0, stream>>>(Osumb, WpbT, bp, out, M, D, D);
}
